// EPD_with_sampling_10213432229974
// MI455X (gfx1250) — compile-verified, matrix operand order checked by reading, not run on hardware
//
#include <hip/hip_runtime.h>
#include <math.h>

#define N_NODES 50000
#define N_EDGES 800000
#define B_GR    8
#define H       128
#define FF      64
#define REPEATS 4
#define OUTD    4

typedef float  v2f   __attribute__((ext_vector_type(2)));
typedef float  v8f   __attribute__((ext_vector_type(8)));
typedef __bf16 v8bf  __attribute__((ext_vector_type(8)));
typedef __bf16 v16bf __attribute__((ext_vector_type(16)));

__device__ __forceinline__ v8f wmma_f32_4(v2f a, v2f b, v8f c) {
    // D = A(16x4,f32) * B(4x16,f32) + C(16x16,f32)
    return __builtin_amdgcn_wmma_f32_16x16x4_f32(false, a, false, b, (short)0, c,
                                                 false, false);
}

__device__ __forceinline__ v8f wmma_bf16_32(v16bf a, v16bf b, v8f c) {
    // D = A(16x32,bf16) * B(32x16,bf16) + C(16x16,f32)
    return __builtin_amdgcn_wmma_f32_16x16x32_bf16(false, a, false, b, (short)0, c,
                                                   false, false);
}

// Gather the per-lane 16 bf16 values of a 32-K chunk: two contiguous 8-element
// (16B) groups. Per the 16-bit operand layout, lane half selects K offsets
// {0..7,16..23} (half=0) vs {8..15,24..31} (half=1) — callers pass p0/p1
// already offset by 8*half.
__device__ __forceinline__ v16bf load_a16(const __bf16* p0, const __bf16* p1) {
    v8bf lo = *reinterpret_cast<const v8bf*>(p0);
    v8bf hi = *reinterpret_cast<const v8bf*>(p1);
    v16bf r;
    #pragma unroll
    for (int i = 0; i < 8; ++i) { r[i] = lo[i]; r[8 + i] = hi[i]; }
    return r;
}

// ---------------------------------------------------------------------------
// One-time transpose+convert of weights: W (Kdim x H, f32) -> Wt (H x Kdim, bf16)
// so the WMMA B-operand per-lane chunks are contiguous 16B loads.
// ---------------------------------------------------------------------------
__global__ void w_transpose_bf16_kernel(const float* __restrict__ W,
                                        __bf16* __restrict__ Wt,
                                        int Kdim) {
    int i = blockIdx.x * blockDim.x + threadIdx.x;
    if (i >= Kdim * H) return;
    int k = i / H, col = i % H;
    Wt[(size_t)col * Kdim + k] = (__bf16)W[(size_t)k * H + col];
}

// ---------------------------------------------------------------------------
// Encoder: h = relu(x_full @ W_enc + b) (f32 + bf16 mirror), normalized
// fourier keys, batch counts
// ---------------------------------------------------------------------------
__global__ void encode_kernel(const float* __restrict__ x,
                              const float* __restrict__ xm,
                              const float* __restrict__ pos,
                              const int*   __restrict__ batch,
                              const float* __restrict__ M,
                              const float* __restrict__ W_enc,
                              const float* __restrict__ b_enc,
                              float*  __restrict__ h,
                              __bf16* __restrict__ hb,
                              float*  __restrict__ nk,
                              float*  __restrict__ cnt,
                              float*  __restrict__ bccnt) {
    int n = blockIdx.x * blockDim.x + threadIdx.x;
    if (n >= N_NODES) return;
    float xv[8];
    for (int k = 0; k < 5; ++k) xv[k] = x[n * 5 + k];
    for (int k = 0; k < 3; ++k) xv[5 + k] = xm[n * 3 + k];
    float bc = xv[7];
    for (int j = 0; j < H; ++j) {
        float s = b_enc[j];
        for (int k = 0; k < 8; ++k) s += xv[k] * W_enc[k * H + j];
        float v = fmaxf(s, 0.f);
        h[(size_t)n * H + j]  = v;
        hb[(size_t)n * H + j] = (__bf16)v;
    }
    float p0 = pos[n * 2 + 0], p1 = pos[n * 2 + 1];
    const float kscale = rsqrtf(2.0f * (float)FF);
    float ss = 0.f;
    for (int f = 0; f < FF; ++f) {
        float ph = p0 * M[f] + p1 * M[FF + f];
        float c = __cosf(ph) * kscale, s = __sinf(ph) * kscale;
        ss += c * c + s * s;
    }
    float inv = rsqrtf(fmaxf(ss, 1e-30f));
    for (int f = 0; f < FF; ++f) {
        float ph = p0 * M[f] + p1 * M[FF + f];
        nk[(size_t)n * H + f]      = __cosf(ph) * kscale * inv;
        nk[(size_t)n * H + FF + f] = __sinf(ph) * kscale * inv;
    }
    int b = batch[n];
    atomicAdd(&cnt[b], 1.0f);
    atomicAdd(&bccnt[b], bc);
}

__global__ void deg_kernel(const int* __restrict__ ei, float* __restrict__ deg) {
    int e = blockIdx.x * blockDim.x + threadIdx.x;
    if (e >= N_EDGES) return;
    atomicAdd(&deg[ei[N_EDGES + e]], 1.0f);
}

// ---------------------------------------------------------------------------
// Segment sums over batch using LDS accumulators (ds_add_f32)
// ---------------------------------------------------------------------------
__global__ void seg_sum_kernel(const float* __restrict__ h,
                               const int*   __restrict__ batch,
                               const float* __restrict__ xm, // nullable weight
                               float* __restrict__ out) {
    __shared__ float sh[B_GR * H];
    int t = threadIdx.x;
    for (int i = t; i < B_GR * H; i += blockDim.x) sh[i] = 0.f;
    __syncthreads();
    int n = blockIdx.x * blockDim.x + t;
    if (n < N_NODES) {
        float w = xm ? xm[n * 3 + 2] : 1.0f;
        if (w != 0.f) {
            int b = batch[n];
            for (int j = 0; j < H; ++j)
                atomicAdd(&sh[b * H + j], w * h[(size_t)n * H + j]);
        }
    }
    __syncthreads();
    for (int i = t; i < B_GR * H; i += blockDim.x) atomicAdd(&out[i], sh[i]);
}

__global__ void finalize_mean_kernel(const float* __restrict__ sums,
                                     const float* __restrict__ cnts,
                                     float*  __restrict__ outf,
                                     __bf16* __restrict__ outb) {
    int i = blockIdx.x * blockDim.x + threadIdx.x;
    if (i >= B_GR * H) return;
    float v = sums[i] / fmaxf(cnts[i / H], 1.0f);
    outf[i] = v;
    outb[i] = (__bf16)v;
}

// ---------------------------------------------------------------------------
// Message GEMM: m = relu([h[src]|h[dst]|edge_attr] @ W_msg + b), scatter to agg
// One wave handles 16 edges x 128 columns via 8 f32 accumulators.
// K=0..255 runs in bf16 WMMA (K=32/instr); K=256..258 (edge_attr) stays f32.
// ---------------------------------------------------------------------------
__global__ void __launch_bounds__(256)
message_kernel(const __bf16* __restrict__ hb,
               const int*    __restrict__ ei,
               const float*  __restrict__ ea,
               const __bf16* __restrict__ Wt,   // 128 x 256 (col-major-K) bf16
               const float*  __restrict__ W,    // original 259x128 f32 (tail rows)
               const float*  __restrict__ bias,
               float* __restrict__ agg) {
    const int lane = threadIdx.x & 31;
    const int wave = threadIdx.x >> 5;
    const int tile = blockIdx.x * 8 + wave;
    const int ntiles = N_EDGES / 16;
    if (tile >= ntiles) return;
    const int half = lane >> 4;
    const int m    = lane & 15;
    const int e    = tile * 16 + m;
    const int ho   = 8 * half; // per-lane K sub-offset within a 32-chunk

    const int srcn = ei[e];
    const int dstn = ei[N_EDGES + e];
    const float a0 = ea[e * 3 + 0];
    const float a1 = ea[e * 3 + 1];
    const float a2 = ea[e * 3 + 2];

    v8f c[8];
    // Tail K=256..258 (edge_attr) pre-accumulated into C in f32 via shuffles.
    #pragma unroll
    for (int t = 0; t < 8; ++t) {
        int col = t * 16 + m;
        float w0 = W[(size_t)256 * H + col];
        float w1 = W[(size_t)257 * H + col];
        float w2 = W[(size_t)258 * H + col];
        #pragma unroll
        for (int v = 0; v < 8; ++v) {
            int er = v + 8 * half; // C row (edge within tile) of this element
            float r0 = __shfl(a0, er, 32);
            float r1 = __shfl(a1, er, 32);
            float r2 = __shfl(a2, er, 32);
            c[t][v] = r0 * w0 + r1 * w1 + r2 * w2;
        }
    }

    const __bf16* asrc = hb + (size_t)srcn * H;
    const __bf16* adst = hb + (size_t)dstn * H;

    // K = 0..127 : h[src] (4 chunks of 32)
    #pragma unroll
    for (int ch = 0; ch < 4; ++ch) {
        int kk = ch * 32;
        v16bf a = load_a16(asrc + kk + ho, asrc + kk + 16 + ho);
        #pragma unroll
        for (int t = 0; t < 8; ++t) {
            int col = t * 16 + m;
            const __bf16* wr = Wt + (size_t)col * 256 + kk;
            v16bf b = load_a16(wr + ho, wr + 16 + ho);
            c[t] = wmma_bf16_32(a, b, c[t]);
        }
    }
    // K = 128..255 : h[dst]
    #pragma unroll
    for (int ch = 0; ch < 4; ++ch) {
        int kk = 128 + ch * 32;
        v16bf a = load_a16(adst + (kk - H) + ho, adst + (kk - H) + 16 + ho);
        #pragma unroll
        for (int t = 0; t < 8; ++t) {
            int col = t * 16 + m;
            const __bf16* wr = Wt + (size_t)col * 256 + kk;
            v16bf b = load_a16(wr + ho, wr + 16 + ho);
            c[t] = wmma_bf16_32(a, b, c[t]);
        }
    }

    // bias + relu + atomic scatter to agg[dst]
    #pragma unroll
    for (int t = 0; t < 8; ++t) {
        int col = t * 16 + m;
        float bj = bias[col];
        #pragma unroll
        for (int v = 0; v < 8; ++v) {
            int er = v + 8 * half;
            float val = fmaxf(c[t][v] + bj, 0.f);
            int drow = __shfl(dstn, er, 32);
            atomicAdd(&agg[(size_t)drow * H + col], val);
        }
    }
}

// ---------------------------------------------------------------------------
// Update GEMM: h_out = h + relu([h|agg/deg|xg[batch]|xbc[batch]] @ W_upd + b)
// One wave handles 16 nodes x 128 columns, K=512 in bf16 WMMA.
// ---------------------------------------------------------------------------
__global__ void __launch_bounds__(256)
update_kernel(const float*  __restrict__ h,     // f32 residual source
              const __bf16* __restrict__ hb,    // bf16 A operand
              const float*  __restrict__ agg,
              const float*  __restrict__ deg,
              const __bf16* __restrict__ xgb,
              const __bf16* __restrict__ xbcb,
              const int*    __restrict__ batch,
              const __bf16* __restrict__ Wt,    // 128 x 512 (col-major-K) bf16
              const float*  __restrict__ bias,
              float*  __restrict__ hout,
              __bf16* __restrict__ hbout) {
    const int lane = threadIdx.x & 31;
    const int wave = threadIdx.x >> 5;
    const int tile = blockIdx.x * 8 + wave;
    const int ntiles = N_NODES / 16;
    if (tile >= ntiles) return;
    const int half = lane >> 4;
    const int m    = lane & 15;
    const int n    = tile * 16 + m;
    const int ho   = 8 * half;

    const int bt = batch[n];
    const float invd = 1.0f / fmaxf(deg[n], 1.0f);
    const __bf16* rh = hb   + (size_t)n  * H;
    const float*  ra = agg  + (size_t)n  * H;
    const __bf16* rg = xgb  + (size_t)bt * H;
    const __bf16* rb = xbcb + (size_t)bt * H;

    v8f c[8];
    #pragma unroll
    for (int t = 0; t < 8; ++t)
        #pragma unroll
        for (int v = 0; v < 8; ++v) c[t][v] = 0.f;

    // K = 0..127 : h (bf16)
    #pragma unroll
    for (int ch = 0; ch < 4; ++ch) {
        int kk = ch * 32;
        v16bf a = load_a16(rh + kk + ho, rh + kk + 16 + ho);
        #pragma unroll
        for (int t = 0; t < 8; ++t) {
            int col = t * 16 + m;
            const __bf16* wr = Wt + (size_t)col * 512 + kk;
            v16bf b = load_a16(wr + ho, wr + 16 + ho);
            c[t] = wmma_bf16_32(a, b, c[t]);
        }
    }
    // K = 128..255 : agg/deg (f32 -> bf16 on the fly)
    #pragma unroll
    for (int ch = 0; ch < 4; ++ch) {
        int kk = 128 + ch * 32;
        const float* p0 = ra + (kk - 128) + ho;
        const float* p1 = ra + (kk - 128) + 16 + ho;
        v16bf a;
        #pragma unroll
        for (int i = 0; i < 8; ++i) {
            a[i]     = (__bf16)(p0[i] * invd);
            a[8 + i] = (__bf16)(p1[i] * invd);
        }
        #pragma unroll
        for (int t = 0; t < 8; ++t) {
            int col = t * 16 + m;
            const __bf16* wr = Wt + (size_t)col * 512 + kk;
            v16bf b = load_a16(wr + ho, wr + 16 + ho);
            c[t] = wmma_bf16_32(a, b, c[t]);
        }
    }
    // K = 256..383 : x_graph[batch]
    #pragma unroll
    for (int ch = 0; ch < 4; ++ch) {
        int kk = 256 + ch * 32;
        v16bf a = load_a16(rg + (kk - 256) + ho, rg + (kk - 256) + 16 + ho);
        #pragma unroll
        for (int t = 0; t < 8; ++t) {
            int col = t * 16 + m;
            const __bf16* wr = Wt + (size_t)col * 512 + kk;
            v16bf b = load_a16(wr + ho, wr + 16 + ho);
            c[t] = wmma_bf16_32(a, b, c[t]);
        }
    }
    // K = 384..511 : x_BC[batch]
    #pragma unroll
    for (int ch = 0; ch < 4; ++ch) {
        int kk = 384 + ch * 32;
        v16bf a = load_a16(rb + (kk - 384) + ho, rb + (kk - 384) + 16 + ho);
        #pragma unroll
        for (int t = 0; t < 8; ++t) {
            int col = t * 16 + m;
            const __bf16* wr = Wt + (size_t)col * 512 + kk;
            v16bf b = load_a16(wr + ho, wr + 16 + ho);
            c[t] = wmma_bf16_32(a, b, c[t]);
        }
    }

    #pragma unroll
    for (int t = 0; t < 8; ++t) {
        int col = t * 16 + m;
        float bj = bias[col];
        #pragma unroll
        for (int v = 0; v < 8; ++v) {
            int nr = tile * 16 + v + 8 * half;
            float val = h[(size_t)nr * H + col] + fmaxf(c[t][v] + bj, 0.f);
            hout[(size_t)nr * H + col]  = val;
            hbout[(size_t)nr * H + col] = (__bf16)val;
        }
    }
}

// ---------------------------------------------------------------------------
// kwv = nk^T @ h  (128 x 128, K = N) in exact f32 WMMA (16x16x4).
// ---------------------------------------------------------------------------
__global__ void __launch_bounds__(32)
kwv_kernel(const float* __restrict__ nk,
           const float* __restrict__ h,
           float* __restrict__ kwv) {
    const int lane = threadIdx.x;
    const int half = lane >> 4;
    const int m    = lane & 15;
    const int ti = blockIdx.x >> 3;
    const int tj = blockIdx.x & 7;
    const int ktotal = N_NODES / 4;
    const int per = (ktotal + gridDim.y - 1) / gridDim.y;
    int k0 = blockIdx.y * per;
    int k1 = k0 + per; if (k1 > ktotal) k1 = ktotal;

    v8f c;
    #pragma unroll
    for (int v = 0; v < 8; ++v) c[v] = 0.f;

    const int acol = ti * 16 + m;
    const int bcol = tj * 16 + m;
    for (int ks = k0; ks < k1; ++ks) {
        int nidx = ks * 4 + 2 * half;
        v2f a, b;
        a[0] = nk[(size_t)(nidx + 0) * H + acol];
        a[1] = nk[(size_t)(nidx + 1) * H + acol];
        b[0] = h[(size_t)(nidx + 0) * H + bcol];
        b[1] = h[(size_t)(nidx + 1) * H + bcol];
        c = wmma_f32_4(a, b, c);
    }
    #pragma unroll
    for (int v = 0; v < 8; ++v) {
        int row = ti * 16 + v + 8 * half;
        atomicAdd(&kwv[(size_t)row * H + tj * 16 + m], c[v]);
    }
}

// ---------------------------------------------------------------------------
// vals_sum = sum_n h[n], ks_sum = sum_n nk[n]  (LDS + global atomics)
// ---------------------------------------------------------------------------
__global__ void reduce_kernel(const float* __restrict__ h,
                              const float* __restrict__ nk,
                              float* __restrict__ vsum,
                              float* __restrict__ kssum) {
    __shared__ float shv[H];
    __shared__ float shk[H];
    int t = threadIdx.x;
    if (t < H) { shv[t] = 0.f; shk[t] = 0.f; }
    __syncthreads();
    int n = blockIdx.x * blockDim.x + t;
    if (n < N_NODES) {
        for (int j = 0; j < H; ++j) {
            atomicAdd(&shv[j], h[(size_t)n * H + j]);
            atomicAdd(&shk[j], nk[(size_t)n * H + j]);
        }
    }
    __syncthreads();
    if (t < H) {
        atomicAdd(&vsum[t], shv[t]);
        atomicAdd(&kssum[t], shk[t]);
    }
}

__global__ void decode_kernel(const float* __restrict__ h,
                              const float* __restrict__ W,
                              const float* __restrict__ bias,
                              float* __restrict__ out) {
    int n = blockIdx.x * blockDim.x + threadIdx.x;
    if (n >= N_NODES) return;
    float s[OUTD];
    for (int o = 0; o < OUTD; ++o) s[o] = bias[o];
    for (int j = 0; j < H; ++j) {
        float hv = h[(size_t)n * H + j];
        for (int o = 0; o < OUTD; ++o) s[o] += hv * W[j * OUTD + o];
    }
    for (int o = 0; o < OUTD; ++o) out[(size_t)(1 + n) * OUTD + o] = s[o];
}

// ---------------------------------------------------------------------------
// Final: q-fourier, enc = (vals_sum + nq@kwv)/(N + nq@ks_sum), U = enc@W_dec+b
// ---------------------------------------------------------------------------
__global__ void final_kernel(const float* __restrict__ sp,
                             const float* __restrict__ M,
                             const float* __restrict__ kwv,
                             const float* __restrict__ vsum,
                             const float* __restrict__ kssum,
                             const float* __restrict__ W_dec,
                             const float* __restrict__ b_dec,
                             float* __restrict__ out) {
    __shared__ float nq[H];
    __shared__ float enc[H];
    __shared__ float red[H];
    int j = threadIdx.x; // 0..127
    float s0 = sp[0], s1 = sp[1];
    int f = j & 63;
    float ph  = s0 * M[f] + s1 * M[FF + f];
    float raw = (j < FF) ? __cosf(ph) : __sinf(ph);
    float key = raw * rsqrtf(2.0f * (float)FF);
    red[j] = key * key;
    __syncthreads();
    for (int st = 64; st > 0; st >>= 1) {
        if (j < st) red[j] += red[j + st];
        __syncthreads();
    }
    float inv = rsqrtf(fmaxf(red[0], 1e-30f));
    __syncthreads();
    nq[j] = key * inv;
    __syncthreads();
    red[j] = nq[j] * kssum[j];
    __syncthreads();
    for (int st = 64; st > 0; st >>= 1) {
        if (j < st) red[j] += red[j + st];
        __syncthreads();
    }
    float denom = (float)N_NODES + red[0];
    float inner = 0.f;
    for (int i = 0; i < H; ++i) inner += nq[i] * kwv[(size_t)i * H + j];
    enc[j] = (vsum[j] + inner) / denom;
    __syncthreads();
    if (j < OUTD) {
        float s = b_dec[j];
        for (int i = 0; i < H; ++i) s += enc[i] * W_dec[i * OUTD + j];
        out[j] = s;
    }
}

// ---------------------------------------------------------------------------
extern "C" void kernel_launch(void* const* d_in, const int* in_sizes, int n_in,
                              void* d_out, int out_size, void* d_ws, size_t ws_size,
                              hipStream_t stream) {
    (void)in_sizes; (void)n_in; (void)out_size; (void)ws_size;
    const float* x     = (const float*)d_in[0];
    const float* xm    = (const float*)d_in[1];
    const int*   ei    = (const int*)  d_in[2];
    const float* ea    = (const float*)d_in[3];
    const float* pos   = (const float*)d_in[4];
    const int*   batch = (const int*)  d_in[5];
    const float* sp    = (const float*)d_in[6];
    const float* fm    = (const float*)d_in[7];
    const float* W_enc = (const float*)d_in[8];
    const float* b_enc = (const float*)d_in[9];
    const float* W_msg = (const float*)d_in[10];
    const float* b_msg = (const float*)d_in[11];
    const float* W_upd = (const float*)d_in[12];
    const float* b_upd = (const float*)d_in[13];
    const float* W_dec = (const float*)d_in[14];
    const float* b_dec = (const float*)d_in[15];
    float* out = (float*)d_out;

    float* ws = (float*)d_ws;
    size_t off = 0;
    float* h0    = ws + off; off += (size_t)N_NODES * H;
    float* h1    = ws + off; off += (size_t)N_NODES * H;
    float* agg   = ws + off; off += (size_t)N_NODES * H;
    float* nk    = ws + off; off += (size_t)N_NODES * H;
    float* deg   = ws + off; off += N_NODES;
    float* cnt   = ws + off; off += B_GR;
    float* bccnt = ws + off; off += B_GR;
    float* gsum  = ws + off; off += B_GR * H;
    float* bcsum = ws + off; off += B_GR * H;
    float* xg    = ws + off; off += B_GR * H;
    float* xbc   = ws + off; off += B_GR * H;
    float* kwv   = ws + off; off += H * H;
    float* vsum  = ws + off; off += H;
    float* kssum = ws + off; off += H;

    // bf16 region (2-byte elements) carved after the f32 region
    __bf16* bb = (__bf16*)(ws + off);
    size_t bo = 0;
    __bf16* hb0  = bb + bo; bo += (size_t)N_NODES * H;
    __bf16* hb1  = bb + bo; bo += (size_t)N_NODES * H;
    __bf16* wtm  = bb + bo; bo += (size_t)H * 256;   // W_msg[0:256] transposed
    __bf16* wtu  = bb + bo; bo += (size_t)H * 512;   // W_upd transposed
    __bf16* xgb  = bb + bo; bo += B_GR * H;
    __bf16* xbcb = bb + bo; bo += B_GR * H;

    const int tb = 256;
    hipMemsetAsync(deg, 0, N_NODES * sizeof(float), stream);
    hipMemsetAsync(cnt, 0, (2 * B_GR + 2 * B_GR * H) * sizeof(float), stream);

    w_transpose_bf16_kernel<<<(256 * H + tb - 1) / tb, tb, 0, stream>>>(W_msg, wtm, 256);
    w_transpose_bf16_kernel<<<(512 * H + tb - 1) / tb, tb, 0, stream>>>(W_upd, wtu, 512);

    encode_kernel<<<(N_NODES + tb - 1) / tb, tb, 0, stream>>>(
        x, xm, pos, batch, fm, W_enc, b_enc, h0, hb0, nk, cnt, bccnt);
    deg_kernel<<<(N_EDGES + tb - 1) / tb, tb, 0, stream>>>(ei, deg);
    seg_sum_kernel<<<(N_NODES + tb - 1) / tb, tb, 0, stream>>>(h0, batch, nullptr, gsum);
    seg_sum_kernel<<<(N_NODES + tb - 1) / tb, tb, 0, stream>>>(h0, batch, xm, bcsum);
    finalize_mean_kernel<<<(B_GR * H + tb - 1) / tb, tb, 0, stream>>>(gsum, cnt, xg, xgb);
    finalize_mean_kernel<<<(B_GR * H + tb - 1) / tb, tb, 0, stream>>>(bcsum, bccnt, xbc, xbcb);

    float*  hc  = h0;  float*  hn  = h1;
    __bf16* hbc = hb0; __bf16* hbn = hb1;
    const int msg_blocks = (N_EDGES / 16 + 7) / 8;
    const int upd_blocks = (N_NODES / 16 + 7) / 8;
    for (int it = 0; it < REPEATS; ++it) {
        hipMemsetAsync(agg, 0, (size_t)N_NODES * H * sizeof(float), stream);
        message_kernel<<<msg_blocks, 256, 0, stream>>>(hbc, ei, ea, wtm, W_msg,
                                                       b_msg, agg);
        update_kernel<<<upd_blocks, 256, 0, stream>>>(hc, hbc, agg, deg, xgb, xbcb,
                                                      batch, wtu, b_upd, hn, hbn);
        hipMemsetAsync(gsum, 0, B_GR * H * sizeof(float), stream);
        seg_sum_kernel<<<(N_NODES + tb - 1) / tb, tb, 0, stream>>>(hn, batch, nullptr, gsum);
        finalize_mean_kernel<<<(B_GR * H + tb - 1) / tb, tb, 0, stream>>>(gsum, cnt, xg, xgb);
        float*  tf = hc;  hc  = hn;  hn  = tf;
        __bf16* tbp = hbc; hbc = hbn; hbn = tbp;
    }

    decode_kernel<<<(N_NODES + tb - 1) / tb, tb, 0, stream>>>(hc, W_dec, b_dec, out);

    hipMemsetAsync(kwv, 0, (H * H + 2 * H) * sizeof(float), stream);
    kwv_kernel<<<dim3(64, 25), 32, 0, stream>>>(nk, hc, kwv);
    reduce_kernel<<<(N_NODES + tb - 1) / tb, tb, 0, stream>>>(hc, nk, vsum, kssum);
    final_kernel<<<1, H, 0, stream>>>(sp, fm, kwv, vsum, kssum, W_dec, b_dec, out);
}
